// SIE_Module_59330678227583
// MI455X (gfx1250) — compile-verified, matrix operand order checked by reading, not run on hardware
//
#include <hip/hip_runtime.h>

// Problem constants (from reference):
//   P=6, B=32, CIN=2, NP=256, L=12, C=32, LAYERS=3, N=1536, PB=192
#define PB_TOT 192
#define CC 32
#define NP 256
#define LL 12
#define RT 384          // GEMM rows = C*L, ordered r = t*32 + c
#define LAYERS 3
#define NTOT 1536

typedef __bf16 bf16_t;
typedef bf16_t bf16x8  __attribute__((ext_vector_type(8)));
typedef bf16_t bf16x16 __attribute__((ext_vector_type(16)));
typedef float  floatx4 __attribute__((ext_vector_type(4)));
typedef float  floatx8 __attribute__((ext_vector_type(8)));

// Build a v16bf WMMA fragment from two contiguous 16-byte chunks
// (ISA 16-bit A/B layout: per lane, K = {base..base+7} and {base+16..base+23}).
static __device__ inline bf16x16 make_frag(const bf16_t* lo, const bf16_t* hi) {
  bf16x8 a = *(const bf16x8*)lo;
  bf16x8 b = *(const bf16x8*)hi;
  bf16x16 r;
#pragma unroll
  for (int i = 0; i < 8; ++i) { r[i] = a[i]; r[i + 8] = b[i]; }
  return r;
}

// ---------------------------------------------------------------------------
// Kernel 1: start conv (CIN=2 -> C=32, 1x1) + ReLU, write bf16
//           Y[pb][r = t*32 + c][n]  (t-major rows so 16-row WMMA tiles have a
//           single t and c-aligned 16-groups -> contiguous M stores later)
// ---------------------------------------------------------------------------
__global__ void prep_y_kernel(const float* __restrict__ node_type,
                              const float* __restrict__ W_start,
                              const float* __restrict__ b_start,
                              bf16_t* __restrict__ Y) {
  __shared__ float w0[CC], w1[CC], bs[CC];
  int tid = threadIdx.x;
  if (tid < CC) {
    w0[tid] = W_start[tid * 2 + 0];
    w1[tid] = W_start[tid * 2 + 1];
    bs[tid] = b_start[tid];
  }
  __syncthreads();
  int pb = blockIdx.x;
  const float* nt = node_type + (size_t)pb * 2 * NP * LL;   // [cin][n][t]
  bf16_t* yout = Y + (size_t)pb * RT * NP;
  for (int idx = tid; idx < RT * NP; idx += blockDim.x) {
    int r = idx >> 8;             // t*32 + c
    int n = idx & (NP - 1);
    int t = r >> 5;
    int c = r & 31;
    float v0 = nt[(size_t)n * LL + t];
    float v1 = nt[(size_t)(NP + n) * LL + t];
    float f = fmaf(w0[c], v0, fmaf(w1[c], v1, bs[c]));
    yout[idx] = (bf16_t)fmaxf(f, 0.0f);
  }
}

// ---------------------------------------------------------------------------
// Kernel 2: S -> bf16 twice: Sb[m][n] = S[m][n] (for x*S^T) and
//           St[m][n] = S[n][m]        (for x*S), via LDS 32x32 transpose.
// ---------------------------------------------------------------------------
__global__ void prep_s_kernel(const float* __restrict__ stg,
                              bf16_t* __restrict__ Sb,
                              bf16_t* __restrict__ St) {
  __shared__ bf16_t tile[32][33];
  int pb  = blockIdx.x >> 6;
  int tix = blockIdx.x & 63;
  int n0 = (tix >> 3) * 32;
  int m0 = (tix & 7) * 32;
  int j  = threadIdx.x & 31;
  int i0 = threadIdx.x >> 5;
  const float* s = stg + (size_t)pb * NP * NP;
  bf16_t* sb = Sb + (size_t)pb * NP * NP;
  bf16_t* st = St + (size_t)pb * NP * NP;
  for (int ii = i0; ii < 32; ii += 8) {
    bf16_t bv = (bf16_t)s[(size_t)(n0 + ii) * NP + m0 + j];
    sb[(size_t)(n0 + ii) * NP + m0 + j] = bv;   // row-major copy
    tile[ii][j] = bv;
  }
  __syncthreads();
  for (int ii = i0; ii < 32; ii += 8) {
    st[(size_t)(m0 + ii) * NP + n0 + j] = tile[j][ii];   // transposed copy
  }
}

// ---------------------------------------------------------------------------
// Kernel 3: batched propagation GEMM (loop-invariant across the 3 layers!)
//   M1[r][m] = sum_n Y[r][n] * S[n][m]   (B column from St, contiguous)
//   M2[r][m] = sum_n Y[r][n] * S[m][n]   (B column from Sb, contiguous)
// Block = (pb, 64-row group): A staged in LDS via async global->LDS copies.
// Each wave owns 4 (m-tile, which) pairs; one B fragment is reused across
// 4 row-tiles (4 live accumulators) -> 0.5 global b128 loads per WMMA.
// D rows are c-contiguous (r = t*32+c), so each lane stores its 8 results as
// ONE b128 into M[which][pb][t][m][c].
// ---------------------------------------------------------------------------
__global__ void gemm_prop_kernel(const bf16_t* __restrict__ Y,
                                 const bf16_t* __restrict__ Sb,
                                 const bf16_t* __restrict__ St,
                                 bf16_t* __restrict__ M) {
  __shared__ bf16_t Alds[64][264];   // 64 rows x 256 K, padded stride (33 KB)
  int pb  = blockIdx.x / 6;
  int rtg = blockIdx.x % 6;          // rows rtg*64 .. +64  (t = 2*rtg, 2*rtg+1)
  int tid = threadIdx.x;
  {
    // Async copy 64x256 bf16 (32 KB): each thread moves 128 B as 8 b128 ops.
    int row = tid >> 2;
    int seg = (tid & 3) * 64;        // bf16 elements
    unsigned long long sbase =
        (unsigned long long)(uintptr_t)(Y + ((size_t)pb * RT + rtg * 64) * NP);
    unsigned vbyte = (unsigned)(row * NP + seg) * 2u;
    unsigned lbyte = (unsigned)(uintptr_t)&Alds[row][seg];
#pragma unroll
    for (int q = 0; q < 8; ++q) {
      asm volatile("global_load_async_to_lds_b128 %0, %1, %2"
                   :
                   : "v"(lbyte + 16u * q), "v"(vbyte + 16u * q), "s"(sbase)
                   : "memory");
    }
    asm volatile("s_wait_asynccnt 0x0" ::: "memory");
  }
  __syncthreads();

  int wave = tid >> 5;
  int lane = tid & 31;
  int lrow = lane & 15;
  int hi8  = (lane >> 4) << 3;       // 0 | 8 per lane half
#pragma unroll
  for (int s = 0; s < 4; ++s) {
    int pair  = wave * 4 + s;        // 32 pairs: 16 m-tiles x {M1, M2}
    int which = pair >> 4;           // 0 -> M1 (St), 1 -> M2 (Sb)
    int mt    = pair & 15;
    const bf16_t* Bbase = (which == 0 ? St : Sb)
                        + ((size_t)pb * NP + mt * 16 + lrow) * NP + hi8;
    floatx8 acc0 = {0.f,0.f,0.f,0.f,0.f,0.f,0.f,0.f};
    floatx8 acc1 = acc0, acc2 = acc0, acc3 = acc0;
#pragma unroll
    for (int k = 0; k < NP; k += 32) {
      bf16x16 bfrag = make_frag(Bbase + k, Bbase + k + 16);
      bf16x16 a0 = make_frag(&Alds[ 0 + lrow][k + hi8], &Alds[ 0 + lrow][k + hi8 + 16]);
      bf16x16 a1 = make_frag(&Alds[16 + lrow][k + hi8], &Alds[16 + lrow][k + hi8 + 16]);
      bf16x16 a2 = make_frag(&Alds[32 + lrow][k + hi8], &Alds[32 + lrow][k + hi8 + 16]);
      bf16x16 a3 = make_frag(&Alds[48 + lrow][k + hi8], &Alds[48 + lrow][k + hi8 + 16]);
      acc0 = __builtin_amdgcn_wmma_f32_16x16x32_bf16(false, a0, false, bfrag, (short)0, acc0, false, false);
      acc1 = __builtin_amdgcn_wmma_f32_16x16x32_bf16(false, a1, false, bfrag, (short)0, acc1, false, false);
      acc2 = __builtin_amdgcn_wmma_f32_16x16x32_bf16(false, a2, false, bfrag, (short)0, acc2, false, false);
      acc3 = __builtin_amdgcn_wmma_f32_16x16x32_bf16(false, a3, false, bfrag, (short)0, acc3, false, false);
    }
    int m = mt * 16 + lrow;
    bf16_t* Mb = M + (size_t)which * ((size_t)PB_TOT * LL * NP * CC);
    floatx8 accs[4] = {acc0, acc1, acc2, acc3};
#pragma unroll
    for (int rtl = 0; rtl < 4; ++rtl) {
      int rglob = rtg * 64 + rtl * 16;
      int t  = rglob >> 5;
      int cb = (rglob & 31) + hi8;           // 0|8|16|24: 16B-aligned run of 8 c
      bf16x8 o8;
#pragma unroll
      for (int v = 0; v < 8; ++v) o8[v] = (bf16_t)accs[rtl][v];
      *(bf16x8*)(Mb + (((size_t)pb * LL + t) * NP + m) * CC + cb) = o8;
    }
  }
}

// ---------------------------------------------------------------------------
// Kernel 4: per-layer channel mix (K=32 WMMA) + bias, assembled in LDS, then
// perm-scattered to out as CONTIGUOUS 144 B runs (9 x global_store_b128).
// Block = 128 threads (4 waves), covers (pb, 16-m block). Each wave owns
// 3 j-tiles; per j-tile the two M fragments are loaded once and reused across
// 3 layers x 2 o-halves (12 WMMAs per 2 fragment loads).
// ---------------------------------------------------------------------------
__global__ void mix_scatter_kernel(const bf16_t* __restrict__ M,
                                   const float* __restrict__ W1,
                                   const float* __restrict__ b1,
                                   const float* __restrict__ W2,
                                   const float* __restrict__ b2,
                                   const int* __restrict__ perm,
                                   float* __restrict__ out) {
  __shared__ bf16_t Wt[LAYERS * 2 * CC * CC];   // [i][which][o][c]  (12 KB)
  __shared__ float  bsum[LAYERS][CC];
  __shared__ float  Hl[CC][16][LAYERS * LL];    // [o][m_local][tt]  (72 KB)
  int tid = threadIdx.x;
  for (int idx = tid; idx < LAYERS * 2 * CC * CC; idx += blockDim.x) {
    int c  = idx & 31;
    int o  = (idx >> 5) & 31;
    int wh = (idx >> 10) & 1;
    int i  = idx >> 11;
    const float* W = wh ? W2 : W1;
    Wt[idx] = (bf16_t)W[((size_t)i * CC + o) * CC + c];
  }
  if (tid < LAYERS * CC) {
    int i = tid >> 5, o = tid & 31;
    bsum[i][o] = b1[i * CC + o] + b2[i * CC + o];
  }
  __syncthreads();
  int pb = blockIdx.x >> 3;
  int mb = blockIdx.x & 7;
  int m0 = mb * 16;
  int wave = tid >> 5;              // 0..3
  int lane = tid & 31;
  int lrow = lane & 15;
  int hi8  = (lane >> 4) << 3;
#pragma unroll
  for (int js = 0; js < 3; ++js) {
    int jt   = wave * 3 + js;       // 0..11  (j = t*16 + m_local)
    int jcol = jt * 16 + lrow;
    int t  = jcol >> 4;
    int ml = jcol & 15;
    bf16x16 bf[2];
#pragma unroll
    for (int wh = 0; wh < 2; ++wh) {
      const bf16_t* bcol = M + (size_t)wh * ((size_t)PB_TOT * LL * NP * CC)
                             + (((size_t)pb * LL + t) * NP + m0 + ml) * CC + hi8;
      bf[wh] = make_frag(bcol, bcol + 16);
    }
#pragma unroll
    for (int i = 0; i < LAYERS; ++i) {
#pragma unroll
      for (int oh = 0; oh < 2; ++oh) {
        floatx8 acc;
#pragma unroll
        for (int v = 0; v < 8; ++v) acc[v] = bsum[i][oh * 16 + hi8 + v];
#pragma unroll
        for (int wh = 0; wh < 2; ++wh) {
          const bf16_t* arow = &Wt[(((i * 2 + wh) * CC) + oh * 16 + lrow) * CC + hi8];
          bf16x16 afrag = make_frag(arow, arow + 16);
          acc = __builtin_amdgcn_wmma_f32_16x16x32_bf16(
              false, afrag, false, bf[wh], (short)0, acc, false, false);
        }
        int tt = i * LL + t;
#pragma unroll
        for (int v = 0; v < 8; ++v) Hl[oh * 16 + hi8 + v][ml][tt] = acc[v];
      }
    }
  }
  __syncthreads();
  int p    = pb >> 5;               // pb = p*32 + b
  int bidx = pb & 31;
  for (int pair = tid; pair < CC * 16; pair += blockDim.x) {
    int o  = pair >> 4;
    int ml = pair & 15;
    int outNode = perm[p * NP + m0 + ml];
    float* dst = out + ((size_t)(bidx * CC + o) * NTOT + outNode) * (LAYERS * LL);
    const floatx4* src = (const floatx4*)&Hl[o][ml][0];
#pragma unroll
    for (int q = 0; q < 9; ++q) ((floatx4*)dst)[q] = src[q];   // 144 B run
  }
}

// ---------------------------------------------------------------------------
extern "C" void kernel_launch(void* const* d_in, const int* in_sizes, int n_in,
                              void* d_out, int out_size, void* d_ws, size_t ws_size,
                              hipStream_t stream) {
  (void)in_sizes; (void)n_in; (void)out_size; (void)ws_size;
  const float* node_type = (const float*)d_in[0];
  const float* stg       = (const float*)d_in[1];
  const float* W_start   = (const float*)d_in[2];
  const float* b_start   = (const float*)d_in[3];
  const float* W1        = (const float*)d_in[4];
  const float* b1        = (const float*)d_in[5];
  const float* W2        = (const float*)d_in[6];
  const float* b2        = (const float*)d_in[7];
  const int*   perm      = (const int*)d_in[8];
  float* out = (float*)d_out;

  // Workspace layout (all 16B-aligned):
  bf16_t* Y  = (bf16_t*)d_ws;                          // 192*384*256  (~37.7 MB)
  bf16_t* Sb = Y  + (size_t)PB_TOT * RT * NP;          // 192*256*256  (~25.2 MB)
  bf16_t* St = Sb + (size_t)PB_TOT * NP * NP;          // 192*256*256  (~25.2 MB)
  bf16_t* M  = St + (size_t)PB_TOT * NP * NP;          // 2*192*12*256*32 (~75.5 MB)

  hipLaunchKernelGGL(prep_y_kernel, dim3(PB_TOT), dim3(256), 0, stream,
                     node_type, W_start, b_start, Y);
  hipLaunchKernelGGL(prep_s_kernel, dim3(PB_TOT * 64), dim3(256), 0, stream,
                     stg, Sb, St);
  hipLaunchKernelGGL(gemm_prop_kernel, dim3(PB_TOT * 6), dim3(256), 0, stream,
                     Y, Sb, St, M);
  hipLaunchKernelGGL(mix_scatter_kernel, dim3(PB_TOT * 8), dim3(128), 0, stream,
                     M, W1, b1, W2, b2, perm, out);
}
